// RARE_69106023793306
// MI455X (gfx1250) — matrix operand order repaired and checked by reading, not compiled, not run on hardware
//
#include <hip/hip_runtime.h>
#include <hip/hip_bf16.h>
#include <math.h>

namespace {
constexpr int kNT = 128, kNB = 256, kNC = 512;
constexpr int kHID = 256, kEMB = 128, kNCLS = 37, kNSTEPS = 25, kPOSE = 256;
constexpr int kFC = kNC + kPOSE;                    // 768 = ctx width
constexpr int kRNN_IN = kNC + kEMB + kPOSE + 576;   // 1472
constexpr int kG = 3 * kHID;                        // 768
}

typedef __bf16 bf16;
typedef __attribute__((ext_vector_type(16))) __bf16 v16bf;
typedef __attribute__((ext_vector_type(8)))  __bf16 v8bf;
typedef __attribute__((ext_vector_type(8)))  float  v8f;

// ---------------------------------------------------------------------------
// Fragment helpers (v_wmma_f32_16x16x32_bf16 layouts):
// A (16x32): lane m=lane&15 row; half=lane>>4:
//   a[0..15] = A[row][kt + half*16 + 0..15]
// B (32x16): lane n=lane&15 col; half selects K 0-15/16-31:
//   b[0..15] = W[col][kt + half*16 + 0..15]   (W row-major (N,K))
// C/D: acc[v] -> row = tm*16 + half*8 + v, col = tn*16 + (lane&15)
// ---------------------------------------------------------------------------
__device__ __forceinline__ v16bf load_a_cvt(const float* p)
{
  const float4* q = reinterpret_cast<const float4*>(p);
  float4 f0 = q[0], f1 = q[1], f2 = q[2], f3 = q[3];
  v16bf a;
  a[0]  = (bf16)f0.x; a[1]  = (bf16)f0.y; a[2]  = (bf16)f0.z; a[3]  = (bf16)f0.w;
  a[4]  = (bf16)f1.x; a[5]  = (bf16)f1.y; a[6]  = (bf16)f1.z; a[7]  = (bf16)f1.w;
  a[8]  = (bf16)f2.x; a[9]  = (bf16)f2.y; a[10] = (bf16)f2.z; a[11] = (bf16)f2.w;
  a[12] = (bf16)f3.x; a[13] = (bf16)f3.y; a[14] = (bf16)f3.z; a[15] = (bf16)f3.w;
  return a;
}

__device__ __forceinline__ v16bf load_b(const bf16* p)
{
  const v8bf* wq = reinterpret_cast<const v8bf*>(p);
  v8bf blo = wq[0], bhi = wq[1];
  v16bf b;
#pragma unroll
  for (int j = 0; j < 8; ++j) { b[j] = blo[j]; b[8 + j] = bhi[j]; }
  return b;
}

// ---------------------------------------------------------------------------
// Strip GEMM: each wave computes a 16(M) x 64(N) strip -> 4 accumulators.
// A loaded+converted once per K-step, shared across the 4 WMMAs.
// Requires N % 64 == 0, M % 16 == 0, K % 32 == 0.
// ---------------------------------------------------------------------------
__global__ void __launch_bounds__(256)
wmma_gemm_n64(const float* __restrict__ A, const bf16* __restrict__ W,
              const float* __restrict__ bias, float* __restrict__ C,
              int K, int lda, int ldw, int ldc, int groupsN, int nTiles)
{
  const int wave = threadIdx.x >> 5;
  const int lane = threadIdx.x & 31;
  const int tile = blockIdx.x * 8 + wave;
  if (tile >= nTiles) return;                 // wave-uniform
  const int tm = tile / groupsN;
  const int tg = tile - tm * groupsN;
  const int half = lane >> 4;
  const int r = lane & 15;

  const float* aBase = A + (size_t)(tm * 16 + r) * lda + half * 16;
  const bf16*  wBase = W + (size_t)(tg * 64 + r) * ldw + half * 16;
  const size_t wSub = (size_t)16 * ldw;       // column-subtile stride in W

  v8f acc[4];
#pragma unroll
  for (int n = 0; n < 4; ++n)
#pragma unroll
    for (int v = 0; v < 8; ++v) acc[n][v] = 0.f;

  for (int kt = 0; kt < K; kt += 32) {
    if (kt + 32 < K) __builtin_prefetch(aBase + kt + 32, 0, 3);
    v16bf a = load_a_cvt(aBase + kt);
    const bf16* wk = wBase + kt;
#pragma unroll
    for (int n = 0; n < 4; ++n) {
      v16bf b = load_b(wk + (size_t)n * wSub);
      acc[n] = __builtin_amdgcn_wmma_f32_16x16x32_bf16(
          /*neg_a=*/false, a, /*neg_b=*/false, b,
          /*c_mod=*/(short)0, acc[n], /*reuse_a=*/false, /*reuse_b=*/false);
    }
  }

#pragma unroll
  for (int n = 0; n < 4; ++n) {
    int col = tg * 64 + n * 16 + r;
    float bv = bias ? bias[col] : 0.0f;
#pragma unroll
    for (int v = 0; v < 8; ++v) {
      int row = tm * 16 + half * 8 + v;
      C[(size_t)row * ldc + col] = acc[n][v] + bv;
    }
  }
}

// ---------------------------------------------------------------------------
// Guarded 16x16-per-wave GEMM for ragged N (generator head, N=37).
// ---------------------------------------------------------------------------
__global__ void __launch_bounds__(256)
wmma_gemm_bias(const float* __restrict__ A, const bf16* __restrict__ W,
               const float* __restrict__ bias, float* __restrict__ C,
               int M, int N, int K, int lda, int ldw, int ldc,
               int tilesN, int nTiles)
{
  const int wave = threadIdx.x >> 5;
  const int lane = threadIdx.x & 31;
  const int tile = blockIdx.x * 8 + wave;
  if (tile >= nTiles) return;                 // wave-uniform
  const int tm = tile / tilesN;
  const int tn = tile - tm * tilesN;
  const int half = lane >> 4;
  const int r = lane & 15;

  const int bcol = tn * 16 + r;
  const bool colOK = (bcol < N);
  const float* aBase = A + (size_t)(tm * 16 + r) * lda + half * 16;
  const bf16*  wBase = colOK ? (W + (size_t)bcol * ldw + half * 16) : W;

  v8f acc = {};
  for (int kt = 0; kt < K; kt += 32) {
    v16bf a = load_a_cvt(aBase + kt);
    v16bf b = load_b(wBase + kt);
    acc = __builtin_amdgcn_wmma_f32_16x16x32_bf16(
        false, a, false, b, (short)0, acc, false, false);
  }

  if (colOK) {
    float bv = bias ? bias[bcol] : 0.0f;
#pragma unroll
    for (int v = 0; v < 8; ++v) {
      int row = tm * 16 + half * 8 + v;
      C[(size_t)row * ldc + bcol] = acc[v] + bv;
    }
  }
}

// ---------------------------------------------------------------------------
// Attention: e[t] = sum_h score_w[h]*tanh(fp[t,b,h]+hp[b,h]); softmax over t.
// One block per b. alpha stored (t, b).
// ---------------------------------------------------------------------------
__global__ void __launch_bounds__(256)
attn_kernel(const float* __restrict__ fp, const float* __restrict__ hp,
            const float* __restrict__ score_w, float* __restrict__ alpha)
{
  __shared__ float s_e[kNT];
  __shared__ float s_r[256];
  const int b = blockIdx.x;
  const int wave = threadIdx.x >> 5, lane = threadIdx.x & 31;
  const float* hpb = hp + (size_t)b * kHID;
  for (int t = wave; t < kNT; t += 8) {
    const float* fpt = fp + ((size_t)t * kNB + b) * kHID;
    float s = 0.f;
    for (int h = lane; h < kHID; h += 32)
      s += score_w[h] * tanhf(fpt[h] + hpb[h]);
#pragma unroll
    for (int m = 16; m > 0; m >>= 1) s += __shfl_xor(s, m, 32);
    if (lane == 0) s_e[t] = s;
  }
  __syncthreads();
  float v = (threadIdx.x < kNT) ? s_e[threadIdx.x] : -3.4e38f;
  s_r[threadIdx.x] = v; __syncthreads();
  for (int off = 128; off > 0; off >>= 1) {
    if (threadIdx.x < off) s_r[threadIdx.x] = fmaxf(s_r[threadIdx.x], s_r[threadIdx.x + off]);
    __syncthreads();
  }
  float mx = s_r[0]; __syncthreads();
  float e = (threadIdx.x < kNT) ? __expf(s_e[threadIdx.x] - mx) : 0.f;
  s_r[threadIdx.x] = e; __syncthreads();
  for (int off = 128; off > 0; off >>= 1) {
    if (threadIdx.x < off) s_r[threadIdx.x] += s_r[threadIdx.x + off];
    __syncthreads();
  }
  float denom = s_r[0];
  if (threadIdx.x < kNT) alpha[(size_t)threadIdx.x * kNB + b] = e / denom;
}

// ctx[b,c] = sum_t alpha[t,b] * (c<512 ? feats[t,b,c] : pose[b,c-512,0,t])
__global__ void __launch_bounds__(256)
ctx_kernel(const float* __restrict__ feats, const float* __restrict__ pose,
           const float* __restrict__ alpha, float* __restrict__ ctx)
{
  __shared__ float s_a[kNT];
  const int b = blockIdx.x;
  if (threadIdx.x < kNT) s_a[threadIdx.x] = alpha[(size_t)threadIdx.x * kNB + b];
  __syncthreads();
  for (int c = threadIdx.x; c < kFC; c += 256) {
    float acc = 0.f;
    if (c < kNC) {
      const float* fb = feats + (size_t)b * kNC + c;
      for (int t = 0; t < kNT; ++t) acc += s_a[t] * fb[(size_t)t * kNB * kNC];
    } else {
      const float* pb = pose + (size_t)b * kPOSE * kNT + (size_t)(c - kNC) * kNT;
      for (int t = 0; t < kNT; ++t) acc += s_a[t] * pb[t];
    }
    ctx[(size_t)b * kFC + c] = acc;
  }
}

// coord[b,j] = sigmoid(ctx[b,:] . pose_w[j,:] + pose_b[j]);  1 block x 256 thr
__global__ void coord_kernel(const float* __restrict__ ctx,
                             const float* __restrict__ pose_w,
                             const float* __restrict__ pose_b,
                             float* __restrict__ coord)
{
  const int b = threadIdx.x;
  const float* cb = ctx + (size_t)b * kFC;
#pragma unroll
  for (int j = 0; j < 4; ++j) {
    float s = pose_b[j];
    const float* w = pose_w + j * kFC;
    for (int c = 0; c < kFC; ++c) s += cb[c] * w[c];
    coord[b * 4 + j] = 1.f / (1.f + __expf(-s));
  }
}

// ROI-align 2x2 for one pyramid level; mutates cumulative coord (as reference).
__device__ __forceinline__ void roi_one(const float* __restrict__ f,
                                        int C, int H, int W,
                                        float* __restrict__ out,
                                        float& c0, float& c1, float& c2, float& c3,
                                        int tid)
{
  // reference: coord(x1,y1,x2,y2) *= scale(h,w,h,w) -- replicated verbatim
  c0 *= (float)H; c1 *= (float)W; c2 *= (float)H; c3 *= (float)W;
  float x1 = c0, y1 = c1, x2 = c2, y2 = c3;
  float bw = fmaxf(x2 - x1, 1.f) * 0.5f;
  float bh = fmaxf(y2 - y1, 1.f) * 0.5f;
  float Hf = (float)H, Wf = (float)W;
  float ly_[2], lx_[2]; int y0i_[2], y1i_[2], x0i_[2], x1i_[2]; bool vy[2], vx[2];
#pragma unroll
  for (int k = 0; k < 2; ++k) {
    float y = y1 + (0.5f + (float)k) * bh;
    vy[k] = (y >= -1.f) && (y <= Hf);
    y = fminf(fmaxf(y, 0.f), Hf - 1.f);
    float y0 = floorf(y);
    y0i_[k] = (int)y0;
    y1i_[k] = (y0i_[k] + 1 < H - 1) ? (y0i_[k] + 1) : (H - 1);
    ly_[k] = y - y0;
    float x = x1 + (0.5f + (float)k) * bw;
    vx[k] = (x >= -1.f) && (x <= Wf);
    x = fminf(fmaxf(x, 0.f), Wf - 1.f);
    float x0 = floorf(x);
    x0i_[k] = (int)x0;
    x1i_[k] = (x0i_[k] + 1 < W - 1) ? (x0i_[k] + 1) : (W - 1);
    lx_[k] = x - x0;
  }
  for (int idx = tid; idx < C * 4; idx += 256) {
    int c = idx >> 2, pt = idx & 3;
    int iy = pt >> 1, ix = pt & 1;
    const float* fc = f + (size_t)c * H * W;
    float ly = ly_[iy], hy = 1.f - ly;
    float lx = lx_[ix], hx = 1.f - lx;
    float v = fc[y0i_[iy] * W + x0i_[ix]] * hy * hx
            + fc[y0i_[iy] * W + x1i_[ix]] * hy * lx
            + fc[y1i_[iy] * W + x0i_[ix]] * ly * hx
            + fc[y1i_[iy] * W + x1i_[ix]] * ly * lx;
    if (!(vy[iy] && vx[ix])) v = 0.f;
    out[idx] = v;
  }
}

// Assemble x = [ctx(768) | emb(128) | crop0(64) | crop1(256) | crop2(256)]
__global__ void __launch_bounds__(256)
assemble_kernel(const float* __restrict__ ctx, const float* __restrict__ coord,
                const int* __restrict__ text, const float* __restrict__ char_emb,
                const float* __restrict__ pyr0, const float* __restrict__ pyr1,
                const float* __restrict__ pyr2, float* __restrict__ x, int step)
{
  const int b = blockIdx.x;
  float* xb = x + (size_t)b * kRNN_IN;
  const float* cb = ctx + (size_t)b * kFC;
  for (int i = threadIdx.x; i < kFC; i += 256) xb[i] = cb[i];
  if (threadIdx.x < kEMB) {
    int tr = (step == 0) ? 0 : (text[b * kNSTEPS + step - 1] + 1);
    xb[kFC + threadIdx.x] = char_emb[(size_t)tr * kEMB + threadIdx.x];
  }
  float c0 = coord[b * 4 + 0], c1 = coord[b * 4 + 1];
  float c2 = coord[b * 4 + 2], c3 = coord[b * 4 + 3];
  // pyramid slabs are f[0] (batch 0) only, as in the reference
  roi_one(pyr0, 16, 16, 128, xb + kFC + kEMB,            c0, c1, c2, c3, threadIdx.x);
  roi_one(pyr1, 64,  8,  64, xb + kFC + kEMB + 64,       c0, c1, c2, c3, threadIdx.x);
  roi_one(pyr2, 64,  4,  65, xb + kFC + kEMB + 64 + 256, c0, c1, c2, c3, threadIdx.x);
}

// GRU gates: r,z,n -> h_new; writes hidden in place and hs_all[b,step,:]
__global__ void __launch_bounds__(256)
gates_kernel(const float* __restrict__ gi, const float* __restrict__ gh,
             float* __restrict__ hidden, float* __restrict__ hs_all, int step)
{
  const int tid = blockIdx.x * 256 + threadIdx.x;
  const int b = tid >> 8, h = tid & 255;
  const float* gib = gi + (size_t)b * kG;
  const float* ghb = gh + (size_t)b * kG;
  float r = 1.f / (1.f + __expf(-(gib[h] + ghb[h])));
  float z = 1.f / (1.f + __expf(-(gib[kHID + h] + ghb[kHID + h])));
  float n = tanhf(gib[2 * kHID + h] + r * ghb[2 * kHID + h]);
  float hold = hidden[(size_t)b * kHID + h];
  float hn = (1.f - z) * n + z * hold;
  hidden[(size_t)b * kHID + h] = hn;
  hs_all[((size_t)b * kNSTEPS + step) * kHID + h] = hn;
}

__global__ void f32_to_bf16_kernel(const float* __restrict__ in,
                                   bf16* __restrict__ out, int n)
{
  int i = blockIdx.x * 256 + threadIdx.x;
  if (i < n) out[i] = (bf16)in[i];
}

__global__ void zero_f32_kernel(float* __restrict__ p, int n)
{
  int i = blockIdx.x * 256 + threadIdx.x;
  if (i < n) p[i] = 0.f;
}

// ---------------------------------------------------------------------------
static void launch_gemm64(const float* A, const bf16* W, const float* bias,
                          float* C, int M, int N, int K,
                          int lda, int ldw, int ldc, hipStream_t s)
{
  int tilesM = M / 16;
  int groupsN = N / 64;
  int nTiles = tilesM * groupsN;
  int blocks = (nTiles + 7) / 8;
  wmma_gemm_n64<<<blocks, 256, 0, s>>>(A, W, bias, C, K, lda, ldw, ldc,
                                       groupsN, nTiles);
}

static void launch_gemm(const float* A, const bf16* W, const float* bias,
                        float* C, int M, int N, int K,
                        int lda, int ldw, int ldc, hipStream_t s)
{
  int tilesM = M / 16;
  int tilesN = (N + 15) / 16;
  int nTiles = tilesM * tilesN;
  int blocks = (nTiles + 7) / 8;
  wmma_gemm_bias<<<blocks, 256, 0, s>>>(A, W, bias, C, M, N, K, lda, ldw, ldc,
                                        tilesN, nTiles);
}

extern "C" void kernel_launch(void* const* d_in, const int* in_sizes, int n_in,
                              void* d_out, int out_size, void* d_ws, size_t ws_size,
                              hipStream_t stream)
{
  (void)in_sizes; (void)n_in; (void)out_size; (void)ws_size;
  const float* feats    = (const float*)d_in[0];
  const float* pose     = (const float*)d_in[1];
  const float* pyr0     = (const float*)d_in[2];
  const float* pyr1     = (const float*)d_in[3];
  const float* pyr2     = (const float*)d_in[4];
  const int*   text     = (const int*)d_in[6];
  const float* i2h_w    = (const float*)d_in[7];
  const float* h2h_w    = (const float*)d_in[8];
  const float* h2h_b    = (const float*)d_in[9];
  const float* score_w  = (const float*)d_in[10];
  const float* pose_w   = (const float*)d_in[11];
  const float* pose_b   = (const float*)d_in[12];
  const float* gru_w_ih = (const float*)d_in[13];
  const float* gru_w_hh = (const float*)d_in[14];
  const float* gru_b_ih = (const float*)d_in[15];
  const float* gru_b_hh = (const float*)d_in[16];
  const float* char_emb = (const float*)d_in[17];
  const float* gen_w    = (const float*)d_in[18];
  const float* gen_b    = (const float*)d_in[19];
  float* out = (float*)d_out;

  char* ws = (char*)d_ws;
  size_t off = 0;
  auto alloc = [&](size_t bytes) -> char* {
    char* p = ws + off;
    off += (bytes + 255) & ~(size_t)255;
    return p;
  };
  bf16* i2h_bf  = (bf16*)alloc((size_t)kHID * kNC * sizeof(bf16));
  bf16* h2h_bf  = (bf16*)alloc((size_t)kHID * kHID * sizeof(bf16));
  bf16* wih_bf  = (bf16*)alloc((size_t)kG * kRNN_IN * sizeof(bf16));
  bf16* whh_bf  = (bf16*)alloc((size_t)kG * kHID * sizeof(bf16));
  bf16* gen_bf  = (bf16*)alloc((size_t)kNCLS * kHID * sizeof(bf16));
  float* feats_proj = (float*)alloc((size_t)kNT * kNB * kHID * sizeof(float));
  float* hidden = (float*)alloc((size_t)kNB * kHID * sizeof(float));
  float* hp     = (float*)alloc((size_t)kNB * kHID * sizeof(float));
  float* alpha  = (float*)alloc((size_t)kNT * kNB * sizeof(float));
  float* ctxb   = (float*)alloc((size_t)kNB * kFC * sizeof(float));
  float* coord  = (float*)alloc((size_t)kNB * 4 * sizeof(float));
  float* xbuf   = (float*)alloc((size_t)kNB * kRNN_IN * sizeof(float));
  float* gi     = (float*)alloc((size_t)kNB * kG * sizeof(float));
  float* gh     = (float*)alloc((size_t)kNB * kG * sizeof(float));
  float* hs_all = (float*)alloc((size_t)kNB * kNSTEPS * kHID * sizeof(float));

  auto cvt = [&](const float* src, bf16* dst, int n) {
    f32_to_bf16_kernel<<<(n + 255) / 256, 256, 0, stream>>>(src, dst, n);
  };
  cvt(i2h_w,    i2h_bf, kHID * kNC);
  cvt(h2h_w,    h2h_bf, kHID * kHID);
  cvt(gru_w_ih, wih_bf, kG * kRNN_IN);
  cvt(gru_w_hh, whh_bf, kG * kHID);
  cvt(gen_w,    gen_bf, kNCLS * kHID);
  zero_f32_kernel<<<(kNB * kHID + 255) / 256, 256, 0, stream>>>(hidden, kNB * kHID);

  // feats_proj = feats(32768x512) * i2h_w^T  (no bias)
  launch_gemm64(feats, i2h_bf, nullptr, feats_proj,
                kNT * kNB, kHID, kNC, kNC, kNC, kHID, stream);

  for (int step = 0; step < kNSTEPS; ++step) {
    // hp = hidden * h2h_w^T + h2h_b
    launch_gemm64(hidden, h2h_bf, h2h_b, hp, kNB, kHID, kHID,
                  kHID, kHID, kHID, stream);
    attn_kernel<<<kNB, 256, 0, stream>>>(feats_proj, hp, score_w, alpha);
    ctx_kernel<<<kNB, 256, 0, stream>>>(feats, pose, alpha, ctxb);
    coord_kernel<<<1, 256, 0, stream>>>(ctxb, pose_w, pose_b, coord);
    assemble_kernel<<<kNB, 256, 0, stream>>>(ctxb, coord, text, char_emb,
                                             pyr0, pyr1, pyr2, xbuf, step);
    // gi = x * W_ih^T + b_ih ; gh = hidden * W_hh^T + b_hh
    launch_gemm64(xbuf,   wih_bf, gru_b_ih, gi, kNB, kG, kRNN_IN,
                  kRNN_IN, kRNN_IN, kG, stream);
    launch_gemm64(hidden, whh_bf, gru_b_hh, gh, kNB, kG, kHID,
                  kHID, kHID, kG, stream);
    gates_kernel<<<kNB, 256, 0, stream>>>(gi, gh, hidden, hs_all, step);
  }

  // probs = hs_all(6400x256) * gen_w^T + gen_b  -> out(6400x37)
  launch_gemm(hs_all, gen_bf, gen_b, out,
              kNB * kNSTEPS, kNCLS, kHID, kHID, kHID, kNCLS, stream);
}
